// SpatialTransformer_50122268344635
// MI455X (gfx1250) — matrix operand order repaired and checked by reading, not compiled, not run on hardware
//
#include <hip/hip_runtime.h>

typedef __attribute__((ext_vector_type(16))) _Float16 v16h;
typedef __attribute__((ext_vector_type(8)))  _Float16 v8h;
typedef __attribute__((ext_vector_type(8)))  float    v8f;

#define T_TOK 4096
#define C_DIM 256

// ---------------------------------------------------------------------------
// f32 -> f16 weight conversion
// ---------------------------------------------------------------------------
__global__ void cvt_f32_f16_kernel(const float* __restrict__ src,
                                   _Float16* __restrict__ dst, int n) {
    int i = blockIdx.x * blockDim.x + threadIdx.x;
    if (i < n) dst[i] = (_Float16)src[i];
}

// ---------------------------------------------------------------------------
// BatchNorm (eval) + transpose [C, T] -> token-major f16 [T, C]
// ---------------------------------------------------------------------------
__global__ __launch_bounds__(256)
void bn_kernel(const float* __restrict__ x, const float* __restrict__ g,
               const float* __restrict__ b, const float* __restrict__ mean,
               const float* __restrict__ var, _Float16* __restrict__ xnH) {
    int c = blockIdx.x;
    float sc = g[c] * rsqrtf(var[c] + 1e-5f);
    float sh = b[c] - mean[c] * sc;
    for (int t = threadIdx.x; t < T_TOK; t += blockDim.x) {
        float v = x[(size_t)c * T_TOK + t];
        xnH[(size_t)t * C_DIM + c] = (_Float16)(v * sc + sh);
    }
}

// ---------------------------------------------------------------------------
// LayerNorm over C=256; one block per token; f16 and/or f32 output
// ---------------------------------------------------------------------------
__global__ __launch_bounds__(256)
void ln_kernel(const float* __restrict__ X, const float* __restrict__ g,
               const float* __restrict__ b, _Float16* outH, float* outF) {
    __shared__ float sbuf[256];
    int t = blockIdx.x, c = threadIdx.x;
    float v = X[(size_t)t * C_DIM + c];
    sbuf[c] = v;
    __syncthreads();
    for (int s = 128; s > 0; s >>= 1) {
        if (c < s) sbuf[c] += sbuf[c + s];
        __syncthreads();
    }
    float mu = sbuf[0] * (1.0f / 256.0f);
    __syncthreads();
    float d = v - mu;
    sbuf[c] = d * d;
    __syncthreads();
    for (int s = 128; s > 0; s >>= 1) {
        if (c < s) sbuf[c] += sbuf[c + s];
        __syncthreads();
    }
    float var = sbuf[0] * (1.0f / 256.0f);
    float y = d * rsqrtf(var + 1e-5f) * g[c] + b[c];
    if (outH) outH[(size_t)t * C_DIM + c] = (_Float16)y;
    if (outF) outF[(size_t)t * C_DIM + c] = y;
}

// ---------------------------------------------------------------------------
// WMMA GEMM: out[M,N] = A_f16[M,Kd] @ W_f16[N,Kd]^T + bias
// One wave per block; each wave computes a 32(M) x 64(N) tile (8 accumulators,
// each B operand reused across two WMMAs), looping Kd in steps of 32 with
// v_wmma_f32_16x16x32_f16.
// Optional epilogue features (null disables): f32 residuals (row-major or
// transposed), exact GELU, f32 / f16 / f32-transposed stores, and an f16
// transposed store outHT for columns n >= ntBase (used to lay V out d-major
// for the attention kernel).
// ---------------------------------------------------------------------------
__global__ __launch_bounds__(32)
void wmma_gemm_kernel(const _Float16* __restrict__ A, const _Float16* __restrict__ W,
                      const float* __restrict__ bias,
                      float* outF, _Float16* outH,
                      _Float16* outHT, int ntBase,
                      float* outT,
                      const float* res1, const float* res2, const float* resT,
                      int M, int N, int Kd, int doGelu) {
    const int lane   = threadIdx.x;
    const int half16 = lane >> 4;     // 0 or 1
    const int l16    = lane & 15;
    const int m0 = blockIdx.x * 32;
    const int n0 = blockIdx.y * 64;

    v8f acc[2][4];
#pragma unroll
    for (int i2 = 0; i2 < 2; ++i2)
#pragma unroll
        for (int j = 0; j < 4; ++j)
#pragma unroll
            for (int r = 0; r < 8; ++r) acc[i2][j][r] = 0.0f;

    // A-operands: lane holds row (m0 + i2*16 + l16), 16 contiguous K values [kb+16*half16..+15]
    const _Float16* aBase0 = A + (size_t)(m0 + l16) * Kd + half16 * 16;
    const _Float16* aBase1 = aBase0 + (size_t)16 * Kd;

    for (int k0 = 0; k0 < Kd; k0 += 32) {
        v16h a0, a1;
        {
            v8h lo0 = *(const v8h*)(aBase0 + k0);
            v8h hi0 = *(const v8h*)(aBase0 + k0 + 8);
            v8h lo1 = *(const v8h*)(aBase1 + k0);
            v8h hi1 = *(const v8h*)(aBase1 + k0 + 8);
#pragma unroll
            for (int i = 0; i < 8; ++i) {
                a0[i] = lo0[i]; a0[8 + i] = hi0[i];
                a1[i] = lo1[i]; a1[8 + i] = hi1[i];
            }
        }
        // speculative prefetch of the next K chunk of A (global_prefetch_b8)
        __builtin_prefetch((const void*)(aBase0 + k0 + 32), 0, 1);
#pragma unroll
        for (int j = 0; j < 4; ++j) {
            // B-operand: lane holds column n0+j*16+l16, 16 contiguous K values
            const int n = n0 + j * 16 + l16;
            v16h bm = *(const v16h*)(W + (size_t)n * Kd + k0 + half16 * 16);
            acc[0][j] = __builtin_amdgcn_wmma_f32_16x16x32_f16(
                false, a0, false, bm, (short)0, acc[0][j], false, false);
            acc[1][j] = __builtin_amdgcn_wmma_f32_16x16x32_f16(
                false, a1, false, bm, (short)0, acc[1][j], false, false);
        }
    }

    // Epilogue. C/D layout: value (m_local = r + 8*half16, n = lane&15)
#pragma unroll
    for (int i2 = 0; i2 < 2; ++i2) {
#pragma unroll
        for (int j = 0; j < 4; ++j) {
            const int n = n0 + j * 16 + l16;
            const float bn = bias ? bias[n] : 0.0f;
#pragma unroll
            for (int r = 0; r < 8; ++r) {
                const int m = m0 + i2 * 16 + r + 8 * half16;
                float v = acc[i2][j][r] + bn;
                if (res1) v += res1[(size_t)m * N + n];
                if (res2) v += res2[(size_t)m * N + n];
                if (resT) v += resT[(size_t)n * M + m];
                if (doGelu) v = 0.5f * v * (1.0f + erff(v * 0.70710678118f));
                if (outF) outF[(size_t)m * N + n] = v;
                if (outHT && n >= ntBase) {
                    outHT[(size_t)(n - ntBase) * M + m] = (_Float16)v;
                } else if (outH) {
                    outH[(size_t)m * N + n] = (_Float16)v;
                }
                if (outT) outT[(size_t)n * M + m] = v;
            }
        }
    }
}

// ---------------------------------------------------------------------------
// Flash attention: one wave handles 16 queries of one head; streams all 4096
// keys in blocks of 32 with online softmax. HD = 32 == WMMA K exactly.
// qk layout: [T, 768] f16 with q at +0, k at +256 (head h at h*32).
// vT layout: [256, T] f16, row = h*32 + d  (d-major => contiguous key runs,
// so the PV B-operand is a single 32B load per lane, same as the K operand).
// Output oH: [T, 256] f16 in (head, hd) channel order.
// ---------------------------------------------------------------------------
__global__ __launch_bounds__(32)
void flash_attn_kernel(const _Float16* __restrict__ qkv,
                       const _Float16* __restrict__ vT,
                       _Float16* __restrict__ oH) {
    __shared__ _Float16 ldsP[16 * 32];
    const int lane   = threadIdx.x;
    const int half16 = lane >> 4;
    const int l16    = lane & 15;
    const int h  = blockIdx.y;
    const int q0 = blockIdx.x * 16;
    const float scale = 0.17677669529663687f;  // 1/sqrt(32)

    // Q A-operand (held in registers for whole kernel)
    v16h qa;
    {
        const _Float16* p = qkv + (size_t)(q0 + l16) * 768 + h * 32 + half16 * 16;
        v8h lo = *(const v8h*)p;
        v8h hi = *(const v8h*)(p + 8);
#pragma unroll
        for (int i = 0; i < 8; ++i) { qa[i] = lo[i]; qa[8 + i] = hi[i]; }
    }

    float mrow[8], lrow[8];
    v8f o0, o1, zf;
#pragma unroll
    for (int r = 0; r < 8; ++r) {
        mrow[r] = -1e30f; lrow[r] = 0.0f;
        o0[r] = 0.0f; o1[r] = 0.0f; zf[r] = 0.0f;
    }

    // V B-operand rows for this lane (fixed for whole kernel)
    const _Float16* vRow0 = vT + (size_t)(h * 32 + l16) * T_TOK;       // d 0..15
    const _Float16* vRow1 = vT + (size_t)(h * 32 + 16 + l16) * T_TOK;  // d 16..31

    for (int kv0 = 0; kv0 < T_TOK; kv0 += 32) {
        // K B-operands: kb0 for keys kv0..kv0+15, kb1 for kv0+16..kv0+31
        v16h kb0 = *(const v16h*)(qkv + (size_t)(kv0 + l16) * 768 + 256 + h * 32 + half16 * 16);
        v16h kb1 = *(const v16h*)(qkv + (size_t)(kv0 + 16 + l16) * 768 + 256 + h * 32 + half16 * 16);

        v8f s0 = __builtin_amdgcn_wmma_f32_16x16x32_f16(false, qa, false, kb0, (short)0, zf, false, false);
        v8f s1 = __builtin_amdgcn_wmma_f32_16x16x32_f16(false, qa, false, kb1, (short)0, zf, false, false);

        // Online softmax. Row m = r + 8*half16, column = lane&15 (+16 for s1).
        float rmax[8], rsum[8], alpha[8];
#pragma unroll
        for (int r = 0; r < 8; ++r) {
            s0[r] *= scale; s1[r] *= scale;
            rmax[r] = fmaxf(s0[r], s1[r]);
        }
#pragma unroll
        for (int msk = 1; msk < 16; msk <<= 1)
#pragma unroll
            for (int r = 0; r < 8; ++r)
                rmax[r] = fmaxf(rmax[r], __shfl_xor(rmax[r], msk, 32));
#pragma unroll
        for (int r = 0; r < 8; ++r) {
            float mn = fmaxf(mrow[r], rmax[r]);
            alpha[r] = __expf(mrow[r] - mn);
            mrow[r] = mn;
            s0[r] = __expf(s0[r] - mn);
            s1[r] = __expf(s1[r] - mn);
            rsum[r] = s0[r] + s1[r];
        }
#pragma unroll
        for (int msk = 1; msk < 16; msk <<= 1)
#pragma unroll
            for (int r = 0; r < 8; ++r)
                rsum[r] += __shfl_xor(rsum[r], msk, 32);
#pragma unroll
        for (int r = 0; r < 8; ++r) {
            lrow[r] = lrow[r] * alpha[r] + rsum[r];
            o0[r] *= alpha[r];
            o1[r] *= alpha[r];
        }

        // Rearrange P from C-layout to A-layout through LDS
        __syncthreads();
#pragma unroll
        for (int r = 0; r < 8; ++r) {
            int m = r + 8 * half16;
            ldsP[m * 32 + l16]      = (_Float16)s0[r];
            ldsP[m * 32 + 16 + l16] = (_Float16)s1[r];
        }
        __syncthreads();
        v16h pa;
        {
            const _Float16* p = ldsP + l16 * 32 + half16 * 16;
#pragma unroll
            for (int i = 0; i < 8; ++i) { pa[i] = p[i]; pa[8 + i] = p[8 + i]; }
        }

        // V B-operands from transposed V: one contiguous 32B load per lane
        v16h vb0 = *(const v16h*)(vRow0 + kv0 + half16 * 16);
        v16h vb1 = *(const v16h*)(vRow1 + kv0 + half16 * 16);

        o0 = __builtin_amdgcn_wmma_f32_16x16x32_f16(false, pa, false, vb0, (short)0, o0, false, false);
        o1 = __builtin_amdgcn_wmma_f32_16x16x32_f16(false, pa, false, vb1, (short)0, o1, false, false);
    }

    // Normalize and write out (channel order = head*32 + d, matching reference)
#pragma unroll
    for (int r = 0; r < 8; ++r) {
        int m = r + 8 * half16;
        int t = q0 + m;
        float inv = 1.0f / lrow[r];
        oH[(size_t)t * C_DIM + h * 32 + l16]      = (_Float16)(o0[r] * inv);
        oH[(size_t)t * C_DIM + h * 32 + 16 + l16] = (_Float16)(o1[r] * inv);
    }
}

// ---------------------------------------------------------------------------
extern "C" void kernel_launch(void* const* d_in, const int* in_sizes, int n_in,
                              void* d_out, int out_size, void* d_ws, size_t ws_size,
                              hipStream_t stream) {
    const float* x        = (const float*)d_in[0];
    const float* bn_gamma = (const float*)d_in[1];
    const float* bn_beta  = (const float*)d_in[2];
    const float* bn_mean  = (const float*)d_in[3];
    const float* bn_var   = (const float*)d_in[4];
    const float* w_in     = (const float*)d_in[5];
    const float* b_in     = (const float*)d_in[6];
    const float* ln1_g    = (const float*)d_in[7];
    const float* ln1_b    = (const float*)d_in[8];
    const float* w_qkv    = (const float*)d_in[9];
    const float* b_qkv    = (const float*)d_in[10];
    const float* w_ap     = (const float*)d_in[11];
    const float* b_ap     = (const float*)d_in[12];
    const float* ln2_g    = (const float*)d_in[13];
    const float* ln2_b    = (const float*)d_in[14];
    const float* w_ff1    = (const float*)d_in[15];
    const float* b_ff1    = (const float*)d_in[16];
    const float* w_ff2    = (const float*)d_in[17];
    const float* b_ff2    = (const float*)d_in[18];
    const float* w_out    = (const float*)d_in[19];
    const float* b_out    = (const float*)d_in[20];

    const int T = T_TOK, C = C_DIM;

    // ---- workspace carve-out (256B aligned slots) ----
    char* ws = (char*)d_ws;
    size_t off = 0;
#define WS_ALLOC(ptr_t, name, bytes)                                \
    ptr_t name = (ptr_t)(ws + off);                                 \
    off = (off + (size_t)(bytes) + 255) & ~(size_t)255;

    WS_ALLOC(_Float16*, wInH,  C * C * 2)
    WS_ALLOC(_Float16*, wQkvH, 3 * C * C * 2)
    WS_ALLOC(_Float16*, wApH,  C * C * 2)
    WS_ALLOC(_Float16*, wFf1H, 4 * C * C * 2)
    WS_ALLOC(_Float16*, wFf2H, 4 * C * C * 2)
    WS_ALLOC(_Float16*, wOutH, C * C * 2)
    WS_ALLOC(_Float16*, xnH,   (size_t)T * C * 2)
    WS_ALLOC(float*,    tokF,  (size_t)T * C * 4)
    WS_ALLOC(_Float16*, yH,    (size_t)T * C * 2)
    WS_ALLOC(float*,    yF,    (size_t)T * C * 4)
    WS_ALLOC(_Float16*, qkvH,  (size_t)T * 3 * C * 2)
    WS_ALLOC(_Float16*, vTH,   (size_t)T * C * 2)
    WS_ALLOC(_Float16*, oHb,   (size_t)T * C * 2)
    WS_ALLOC(float*,    tok2F, (size_t)T * C * 4)
    WS_ALLOC(_Float16*, zH,    (size_t)T * C * 2)
    WS_ALLOC(_Float16*, gH,    (size_t)T * 4 * C * 2)
    WS_ALLOC(_Float16*, tok3H, (size_t)T * C * 2)
#undef WS_ALLOC

    // ---- weight conversion f32 -> f16 ----
    cvt_f32_f16_kernel<<<(C * C + 255) / 256, 256, 0, stream>>>(w_in, wInH, C * C);
    cvt_f32_f16_kernel<<<(3 * C * C + 255) / 256, 256, 0, stream>>>(w_qkv, wQkvH, 3 * C * C);
    cvt_f32_f16_kernel<<<(C * C + 255) / 256, 256, 0, stream>>>(w_ap, wApH, C * C);
    cvt_f32_f16_kernel<<<(4 * C * C + 255) / 256, 256, 0, stream>>>(w_ff1, wFf1H, 4 * C * C);
    cvt_f32_f16_kernel<<<(4 * C * C + 255) / 256, 256, 0, stream>>>(w_ff2, wFf2H, 4 * C * C);
    cvt_f32_f16_kernel<<<(C * C + 255) / 256, 256, 0, stream>>>(w_out, wOutH, C * C);

    // ---- BatchNorm + transpose to token-major f16 ----
    bn_kernel<<<C, 256, 0, stream>>>(x, bn_gamma, bn_beta, bn_mean, bn_var, xnH);

    // ---- proj_in: tok = xn @ w_in^T + b_in  (f32) ----
    wmma_gemm_kernel<<<dim3(T / 32, C / 64), 32, 0, stream>>>(
        xnH, wInH, b_in, tokF, nullptr, nullptr, 0, nullptr,
        nullptr, nullptr, nullptr, T, C, C, 0);

    // ---- LN1 -> y (f16 for gemm, f32 for residual) ----
    ln_kernel<<<T, 256, 0, stream>>>(tokF, ln1_g, ln1_b, yH, yF);

    // ---- qkv = y @ w_qkv^T + b_qkv ; q,k stored [T,768], v stored d-major ----
    wmma_gemm_kernel<<<dim3(T / 32, (3 * C) / 64), 32, 0, stream>>>(
        yH, wQkvH, b_qkv, nullptr, qkvH, vTH, 2 * C, nullptr,
        nullptr, nullptr, nullptr, T, 3 * C, C, 0);

    // ---- flash attention ----
    flash_attn_kernel<<<dim3(T / 16, 8), 32, 0, stream>>>(qkvH, vTH, oHb);

    // ---- attn proj + residuals: tok2 = o @ w_ap^T + b_ap + y + tok ----
    wmma_gemm_kernel<<<dim3(T / 32, C / 64), 32, 0, stream>>>(
        oHb, wApH, b_ap, tok2F, nullptr, nullptr, 0, nullptr,
        yF, tokF, nullptr, T, C, C, 0);

    // ---- LN2 -> z (f16) ----
    ln_kernel<<<T, 256, 0, stream>>>(tok2F, ln2_g, ln2_b, zH, nullptr);

    // ---- ff1 + exact GELU: g = gelu(z @ w_ff1^T + b_ff1)  (f16) ----
    wmma_gemm_kernel<<<dim3(T / 32, (4 * C) / 64), 32, 0, stream>>>(
        zH, wFf1H, b_ff1, nullptr, gH, nullptr, 0, nullptr,
        nullptr, nullptr, nullptr, T, 4 * C, C, 1);

    // ---- ff2 + residual: tok3 = g @ w_ff2^T + b_ff2 + tok2  (f16) ----
    wmma_gemm_kernel<<<dim3(T / 32, C / 64), 32, 0, stream>>>(
        gH, wFf2H, b_ff2, nullptr, tok3H, nullptr, 0, nullptr,
        tok2F, nullptr, nullptr, T, C, 4 * C, 0);

    // ---- proj_out (transposed store to [C,H,W]) + outer residual x_in ----
    wmma_gemm_kernel<<<dim3(T / 32, C / 64), 32, 0, stream>>>(
        tok3H, wOutH, b_out, nullptr, nullptr, nullptr, 0, (float*)d_out,
        nullptr, nullptr, x, T, C, C, 0);
}
